// Decoder_23381801959676
// MI455X (gfx1250) — matrix operand order repaired and checked by reading, not compiled, not run on hardware
//
#include <hip/hip_runtime.h>
#include <hip/hip_bf16.h>

// ---------------------------------------------------------------------------
// Problem constants (from reference)
// ---------------------------------------------------------------------------
#define SOS_CODE 1
#define VOCAB 1024
#define EMBED 256
#define UNITS 256
#define BATCH 128
#define SEQ   512

typedef __bf16 bf16_t;
typedef __attribute__((ext_vector_type(16))) __bf16 v16bf;
typedef __attribute__((ext_vector_type(8)))  float  v8f;

// ---------------------------------------------------------------------------
// CDNA5 transcendentals
// ---------------------------------------------------------------------------
__device__ __forceinline__ float fast_tanh(float x) {
#if __has_builtin(__builtin_amdgcn_tanhf)
    return __builtin_amdgcn_tanhf(x);     // v_tanh_f32
#else
    return tanhf(x);
#endif
}
__device__ __forceinline__ float fast_sigm(float x) {
    return 0.5f + 0.5f * fast_tanh(0.5f * x);   // exact identity, rides tanh pipe
}

// ---------------------------------------------------------------------------
// WMMA core (bf16 16x16x32, f32 accumulate)
// ---------------------------------------------------------------------------
__device__ __forceinline__ v8f wmma_bf16(v16bf a, v16bf b, v8f c) {
    return __builtin_amdgcn_wmma_f32_16x16x32_bf16(
        /*neg_a=*/false, a, /*neg_b=*/false, b,
        /*c_mod=*/(short)0, c, /*reuse_a=*/false, /*reuse_b=*/false);
}

// Fragment-major packed layouts: frag = tile index; each frag is 32 lanes x 16 bf16
// (32 contiguous bytes per lane -> 2x global_load_b128, no shuffles).
__device__ __forceinline__ v16bf load_frag(const bf16_t* __restrict__ P,
                                           int frag, int lane) {
    return *reinterpret_cast<const v16bf*>(P + ((size_t)frag * 32 + lane) * 16);
}

// A-fragment element mapping (16-bit A 16x32, ISA 7.12.2):
//   lane = m%16 + 16*hi ; k = (e<8) ? e+8*hi : e+8+8*hi
__device__ __forceinline__ int a_k_of(int e, int hi) {
    return (e < 8) ? (e + 8 * hi) : (e + 8 + 8 * hi);
}
// inverse: given k%32 -> (hi, e)
__device__ __forceinline__ void a_pos_of_k(int kk, int& hi, int& e) {
    hi = (kk >> 3) & 1;                              // 8-15 / 24-31 -> upper half
    e  = (kk < 8) ? kk : (kk < 16 ? kk - 8 : (kk < 24 ? kk - 8 : kk - 16));
}

// ---------------------------------------------------------------------------
// Pack f32 row-major A [M x K] into A-fragment-major bf16.
// frag = mtile*(K/32)+ktile ; dst[(frag*32+lane)*16+e] = A[m][k]
// ---------------------------------------------------------------------------
__global__ __launch_bounds__(256) void pack_a_kernel(const float* __restrict__ src,
                                                     bf16_t* __restrict__ dst,
                                                     int M, int K) {
    const int idx = blockIdx.x * 256 + threadIdx.x;
    if (idx >= M * K) return;
    const int e    = idx & 15;
    const int lane = (idx >> 4) & 31;
    const int frag = idx >> 9;
    const int KT   = K >> 5;
    const int mt   = frag / KT;
    const int kt   = frag - mt * KT;
    const int m    = mt * 16 + (lane & 15);
    const int k    = kt * 32 + a_k_of(e, lane >> 4);
    dst[idx] = (bf16_t)src[(size_t)m * K + k];
}

// ---------------------------------------------------------------------------
// Pack f32 row-major B [K x N] into B-fragment-major bf16.
// frag = ntile*(K/32)+ktile ; lane holds n = nt*16+(lane&15), k = kt*32+a_k_of(e, lane>>4)
// ---------------------------------------------------------------------------
__global__ __launch_bounds__(256) void pack_b_kernel(const float* __restrict__ src,
                                                     bf16_t* __restrict__ dst,
                                                     int K, int N) {
    const int idx = blockIdx.x * 256 + threadIdx.x;
    if (idx >= K * N) return;
    const int e    = idx & 15;
    const int lane = (idx >> 4) & 31;
    const int frag = idx >> 9;
    const int KT   = K >> 5;
    const int nt   = frag / KT;
    const int kt   = frag - nt * KT;
    const int n    = nt * 16 + (lane & 15);
    const int k    = kt * 32 + a_k_of(e, lane >> 4);
    dst[idx] = (bf16_t)src[(size_t)k * N + n];
}

// ---------------------------------------------------------------------------
// Pack initial h0 [BATCH x UNITS] f32 into A-fragment-major packed h buffer
// h_pk layout: [btile(8)][ktile(8)][lane(32)][e(16)]
// ---------------------------------------------------------------------------
__global__ __launch_bounds__(256) void pack_h0_kernel(const float* __restrict__ h0,
                                                      bf16_t* __restrict__ h_pk) {
    const int idx = blockIdx.x * 256 + threadIdx.x;   // BATCH*UNITS = 32768
    if (idx >= BATCH * UNITS) return;
    const int b = idx >> 8, u = idx & 255;
    const int btile = b >> 4, row = b & 15, kt = u >> 5, kk = u & 31;
    int hi, e;
    a_pos_of_k(kk, hi, e);
    const int lane = row + (hi << 4);
    h_pk[(((size_t)btile * 8 + kt) * 32 + lane) * 16 + e] = (bf16_t)h0[idx];
}

// ---------------------------------------------------------------------------
// zx[j][n] = emb[token_j] @ kernel + bias, j=0 -> SOS (t==0), j=1 -> token 0 (t>0)
// ---------------------------------------------------------------------------
__global__ __launch_bounds__(128) void zx_kernel(const float* __restrict__ emb,
                                                 const float* __restrict__ kern,
                                                 const float* __restrict__ bias,
                                                 float* __restrict__ zx) {
    const int col = blockIdx.x * 128 + threadIdx.x;      // 0..1023
    const int tok = (blockIdx.y == 0) ? SOS_CODE : 0;
    const float* e = emb + (size_t)tok * EMBED;
    float acc = bias[col];
#pragma unroll 8
    for (int k = 0; k < EMBED; ++k) acc += e[k] * kern[(size_t)k * (4 * UNITS) + col];
    zx[(size_t)blockIdx.y * (4 * UNITS) + col] = acc;
}

// ---------------------------------------------------------------------------
// w1_enc = enc_outputs @ W1 + b1   (M = B*S = 65536, N = 256, K = 256)
// one wave -> 1 mtile x 4 ntiles (A fragments preloaded, reused 4x)
// ---------------------------------------------------------------------------
__global__ __launch_bounds__(256) void w1enc_kernel(const bf16_t* __restrict__ encA,
                                                    const bf16_t* __restrict__ W1_pk,
                                                    const float* __restrict__ b1,
                                                    float* __restrict__ w1enc) {
    const int lane = threadIdx.x & 31;
    const int wave = threadIdx.x >> 5;
    const int g    = blockIdx.x * 8 + wave;     // 0..16383
    const int mt   = g >> 2;
    const int ntg  = (g & 3) * 4;
    const int m0   = mt * 16;

    v16bf afrag[8];
#pragma unroll
    for (int kt = 0; kt < 8; ++kt) afrag[kt] = load_frag(encA, mt * 8 + kt, lane);

    const int nn = lane & 15;
    const int mm = (lane >> 4) << 3;
#pragma unroll
    for (int j = 0; j < 4; ++j) {
        const int nt = ntg + j;
        v8f acc;
        const float binit = b1[nt * 16 + nn];
#pragma unroll
        for (int r = 0; r < 8; ++r) acc[r] = binit;
#pragma unroll
        for (int kt = 0; kt < 8; ++kt)
            acc = wmma_bf16(afrag[kt], load_frag(W1_pk, nt * 8 + kt, lane), acc);
#pragma unroll
        for (int r = 0; r < 8; ++r)
            w1enc[(size_t)(m0 + mm + r) * UNITS + nt * 16 + nn] = acc[r];
    }
}

// ---------------------------------------------------------------------------
// One LSTM time step (fused): z = zx + h@rec_kernel ; gates -> h,c ;
// w2_dec_t = h_new @ W2 + b2  -> w2_all[t]
// grid = 8 blocks (one 16-row batch tile each), 512 threads = 16 waves
// h state lives in A-fragment-packed bf16 (global h_pk + LDS hpack)
// ---------------------------------------------------------------------------
__global__ __launch_bounds__(512) void lstm_step_kernel(const bf16_t* __restrict__ rec_pk,
                                                        const bf16_t* __restrict__ W2_pk,
                                                        const float* __restrict__ zx,
                                                        const float* __restrict__ b2,
                                                        bf16_t* __restrict__ h_pk,
                                                        float* __restrict__ c,
                                                        float* __restrict__ w2_all,
                                                        int t) {
    __shared__ float  zbuf[16][4 * UNITS];        // 64 KB
    __shared__ bf16_t hpack[8 * 32 * 16];         // 8 KB, A-fragment layout

    const int lane = threadIdx.x & 31;
    const int wave = threadIdx.x >> 5;            // 0..15
    const int bt   = blockIdx.x;                  // batch tile 0..7
    const int m0   = bt * 16;
    const float* zxrow = zx + ((t == 0) ? 0 : 4 * UNITS);

    // ---- Phase A: preload h fragments once, compute 4 z tiles / wave ----
    v16bf afrag[8];
#pragma unroll
    for (int kt = 0; kt < 8; ++kt) afrag[kt] = load_frag(h_pk, bt * 8 + kt, lane);

    const int nn = lane & 15;
    const int mm = (lane >> 4) << 3;
#pragma unroll
    for (int tt = 0; tt < 4; ++tt) {
        const int nt = wave * 4 + tt;             // 0..63 column tile of z
        v8f acc;
        const float zinit = zxrow[nt * 16 + nn];
#pragma unroll
        for (int r = 0; r < 8; ++r) acc[r] = zinit;
#pragma unroll
        for (int kt = 0; kt < 8; ++kt)
            acc = wmma_bf16(afrag[kt], load_frag(rec_pk, nt * 8 + kt, lane), acc);
#pragma unroll
        for (int r = 0; r < 8; ++r) zbuf[mm + r][nt * 16 + nn] = acc[r];
    }
    __syncthreads();

    // ---- Phase B: gates (Keras order i,f,g,o); write h in packed layout ----
    for (int i = threadIdx.x; i < 16 * UNITS; i += 512) {
        const int row = i >> 8;          // 0..15
        const int u   = i & 255;
        const float zi = zbuf[row][u];
        const float zf = zbuf[row][UNITS + u];
        const float zg = zbuf[row][2 * UNITS + u];
        const float zo = zbuf[row][3 * UNITS + u];
        const size_t gidx = (size_t)(m0 + row) * UNITS + u;
        const float cold = c[gidx];
        const float cnew = fast_sigm(zf) * cold + fast_sigm(zi) * fast_tanh(zg);
        const float hnew = fast_sigm(zo) * fast_tanh(cnew);
        c[gidx] = cnew;
        const int kt = u >> 5, kk = u & 31;
        int hi, e;
        a_pos_of_k(kk, hi, e);
        const int pl  = row + (hi << 4);
        const bf16_t hb = (bf16_t)hnew;
        const size_t poff = ((size_t)kt * 32 + pl) * 16 + e;
        hpack[poff] = hb;                                   // for phase C
        h_pk[(size_t)bt * (8 * 32 * 16) + poff] = hb;       // for next step
    }
    __syncthreads();

    // ---- Phase C: w2_dec tile (one 16x16 tile per wave, K = 256) ----
    {
        const int nt = wave;
        v8f acc;
        const float binit = b2[nt * 16 + nn];
#pragma unroll
        for (int r = 0; r < 8; ++r) acc[r] = binit;
#pragma unroll
        for (int kt = 0; kt < 8; ++kt) {
            const v16bf a = *reinterpret_cast<const v16bf*>(hpack + ((size_t)kt * 32 + lane) * 16);
            acc = wmma_bf16(a, load_frag(W2_pk, nt * 8 + kt, lane), acc);
        }
        float* dst = w2_all + ((size_t)t * BATCH + m0) * UNITS;
#pragma unroll
        for (int r = 0; r < 8; ++r) dst[(size_t)(mm + r) * UNITS + nt * 16 + nn] = acc[r];
    }
}

// ---------------------------------------------------------------------------
// Attention + fused softmax.
// block = (b, t) pair; 256 threads (8 waves). Each wave owns a stripe of s;
// each lane covers u = [8*lane, 8*lane+8). logits live only in LDS.
// ---------------------------------------------------------------------------
__global__ __launch_bounds__(256) void attn_kernel(const float* __restrict__ w1enc,
                                                   const float* __restrict__ w2_all,
                                                   const float* __restrict__ V,
                                                   const float* __restrict__ bv,
                                                   float* __restrict__ out) {
    const int b = blockIdx.x >> 9;      // consecutive blocks share b -> L2 reuse
    const int t = blockIdx.x & 511;

    __shared__ float w2row[UNITS];
    __shared__ float Vl[UNITS];
    __shared__ float logits[SEQ];
    __shared__ float sred[8];

    const int tid  = threadIdx.x;
    const int lane = tid & 31;
    const int wave = tid >> 5;

    w2row[tid] = w2_all[((size_t)t * BATCH + b) * UNITS + tid];
    Vl[tid]    = V[tid];
    __syncthreads();

    const float bvv = bv[0];
    const float* base = w1enc + (size_t)b * SEQ * UNITS;

    for (int s = wave; s < SEQ; s += 8) {
        const float* wrow = base + (size_t)s * UNITS + lane * 8;
        if (s + 8 < SEQ)
            __builtin_prefetch(base + (size_t)(s + 8) * UNITS + lane * 8, 0, 0);
        const float4 x0 = *reinterpret_cast<const float4*>(wrow);
        const float4 x1 = *reinterpret_cast<const float4*>(wrow + 4);
        const int u = lane * 8;
        float p;
        p  = fast_tanh(x0.x + w2row[u + 0]) * Vl[u + 0];
        p += fast_tanh(x0.y + w2row[u + 1]) * Vl[u + 1];
        p += fast_tanh(x0.z + w2row[u + 2]) * Vl[u + 2];
        p += fast_tanh(x0.w + w2row[u + 3]) * Vl[u + 3];
        p += fast_tanh(x1.x + w2row[u + 4]) * Vl[u + 4];
        p += fast_tanh(x1.y + w2row[u + 5]) * Vl[u + 5];
        p += fast_tanh(x1.z + w2row[u + 6]) * Vl[u + 6];
        p += fast_tanh(x1.w + w2row[u + 7]) * Vl[u + 7];
#pragma unroll
        for (int off = 16; off > 0; off >>= 1) p += __shfl_xor(p, off, 32);
        if (lane == 0) logits[s] = p + bvv;
    }
    __syncthreads();

    // softmax over s (512 values, 256 threads -> 2 each)
    const float l0 = logits[tid];
    const float l1 = logits[tid + 256];
    float m = fmaxf(l0, l1);
#pragma unroll
    for (int off = 16; off > 0; off >>= 1) m = fmaxf(m, __shfl_xor(m, off, 32));
    if (lane == 0) sred[wave] = m;
    __syncthreads();
    float maxv = sred[0];
#pragma unroll
    for (int i = 1; i < 8; ++i) maxv = fmaxf(maxv, sred[i]);

    const float e0 = __expf(l0 - maxv);
    const float e1 = __expf(l1 - maxv);
    float psum = e0 + e1;
#pragma unroll
    for (int off = 16; off > 0; off >>= 1) psum += __shfl_xor(psum, off, 32);
    __syncthreads();
    if (lane == 0) sred[wave] = psum;
    __syncthreads();
    float tot = sred[0];
#pragma unroll
    for (int i = 1; i < 8; ++i) tot += sred[i];
    const float inv = 1.0f / tot;

    float* orow = out + ((size_t)t * BATCH + b) * SEQ;
    orow[tid]       = e0 * inv;
    orow[tid + 256] = e1 * inv;
}

// ---------------------------------------------------------------------------
// Host-side launcher
// ---------------------------------------------------------------------------
extern "C" void kernel_launch(void* const* d_in, const int* in_sizes, int n_in,
                              void* d_out, int out_size, void* d_ws, size_t ws_size,
                              hipStream_t stream) {
    (void)in_sizes; (void)n_in; (void)out_size; (void)ws_size;

    const float* dec_h   = (const float*)d_in[1];
    const float* dec_c   = (const float*)d_in[2];
    const float* enc     = (const float*)d_in[3];
    const float* emb     = (const float*)d_in[4];
    const float* kern    = (const float*)d_in[5];
    const float* reck    = (const float*)d_in[6];
    const float* bias    = (const float*)d_in[7];
    const float* W1      = (const float*)d_in[8];
    const float* b1      = (const float*)d_in[9];
    const float* W2      = (const float*)d_in[10];
    const float* b2      = (const float*)d_in[11];
    const float* V       = (const float*)d_in[12];
    const float* bv      = (const float*)d_in[13];
    float*       out     = (float*)d_out;

    // workspace carve-up (256B aligned slots)
    char* w = (char*)d_ws;
    size_t off = 0;
    auto carve = [&](size_t bytes) -> char* {
        char* p = w + off;
        off = (off + bytes + 255) & ~(size_t)255;
        return p;
    };
    bf16_t* rec_pk = (bf16_t*)carve((size_t)UNITS * 4 * UNITS * sizeof(bf16_t));
    bf16_t* W1_pk  = (bf16_t*)carve((size_t)UNITS * UNITS * sizeof(bf16_t));
    bf16_t* W2_pk  = (bf16_t*)carve((size_t)UNITS * UNITS * sizeof(bf16_t));
    bf16_t* encA   = (bf16_t*)carve((size_t)BATCH * SEQ * UNITS * sizeof(bf16_t));
    bf16_t* h_pk   = (bf16_t*)carve((size_t)BATCH * UNITS * sizeof(bf16_t));
    float*  zx     = (float*)carve(2 * 4 * UNITS * sizeof(float));
    float*  cbuf   = (float*)carve((size_t)BATCH * UNITS * sizeof(float));
    float*  w1enc  = (float*)carve((size_t)BATCH * SEQ * UNITS * sizeof(float));
    float*  w2all  = (float*)carve((size_t)SEQ * BATCH * UNITS * sizeof(float));

    // 1) pack weights / activations into WMMA fragment-major bf16
    {
        int n = UNITS * 4 * UNITS;   // rec_kernel [256 x 1024] as B
        pack_b_kernel<<<(n + 255) / 256, 256, 0, stream>>>(reck, rec_pk, UNITS, 4 * UNITS);
    }
    {
        int n = UNITS * UNITS;
        pack_b_kernel<<<(n + 255) / 256, 256, 0, stream>>>(W1, W1_pk, UNITS, UNITS);
        pack_b_kernel<<<(n + 255) / 256, 256, 0, stream>>>(W2, W2_pk, UNITS, UNITS);
    }
    {
        int n = BATCH * SEQ * UNITS; // enc_outputs [65536 x 256] as A
        pack_a_kernel<<<(n + 255) / 256, 256, 0, stream>>>(enc, encA, BATCH * SEQ, UNITS);
    }

    // 2) constant-input precompute: zx[0]=emb[SOS]@kernel+bias, zx[1]=emb[0]@kernel+bias
    zx_kernel<<<dim3(8, 2), 128, 0, stream>>>(emb, kern, bias, zx);

    // 3) initial LSTM state: h packed to fragment layout, c copied as f32
    {
        int n = BATCH * UNITS;
        pack_h0_kernel<<<(n + 255) / 256, 256, 0, stream>>>(dec_h, h_pk);
    }
    hipMemcpyAsync(cbuf, dec_c, (size_t)BATCH * UNITS * sizeof(float),
                   hipMemcpyDeviceToDevice, stream);

    // 4) w1_enc = enc @ W1 + b1  (65536 x 256 GEMM, WMMA)
    w1enc_kernel<<<2048, 256, 0, stream>>>(encA, W1_pk, b1, w1enc);

    // 5) sequential LSTM recurrence (fused z / gates / W2 per step)
    for (int t = 0; t < SEQ; ++t)
        lstm_step_kernel<<<8, 512, 0, stream>>>(rec_pk, W2_pk, zx, b2,
                                                h_pk, cbuf, w2all, t);

    // 6) attention + fused softmax over all (t, b)
    attn_kernel<<<BATCH * SEQ, 256, 0, stream>>>(w1enc, w2all, V, bv, out);
}
